// PointNet2_70428873720161
// MI455X (gfx1250) — matrix operand order repaired and checked by reading, not compiled, not run on hardware
//
#include <hip/hip_runtime.h>
#include <hip/hip_bf16.h>

// ---------------------------------------------------------------------------
// PointNet++ segmentation backbone for MI455X (gfx1250, wave32, WMMA).
// All 1x1-conv einsums run as bf16 WMMA 16x16x32 GEMMs with f32 accumulation.
// ---------------------------------------------------------------------------

typedef __attribute__((ext_vector_type(16))) __bf16 bf16x16;
typedef __attribute__((ext_vector_type(8)))  float  f32x8;

#define GWAVES 4   // waves (16-row m-tiles) per GEMM block

// ---------------------------- utility kernels ------------------------------

// (B,C,N) -> (B,N,C)
__global__ void k_transpose_cn_nc(const float* __restrict__ in, float* __restrict__ out,
                                  int N, int C, long total) {
  long i = (long)blockIdx.x * blockDim.x + threadIdx.x;
  if (i >= total) return;
  long b = i / ((long)N * C);
  long r = i - b * (long)N * C;
  int n = (int)(r / C), c = (int)(r % C);
  out[i] = in[((size_t)b * C + c) * N + n];
}

// pack f32 weight [O,C] into zero-padded bf16 [Op,Cp]
__global__ void k_pack_w(const float* __restrict__ W, __bf16* __restrict__ Wp,
                         int O, int C, int Cp, long total) {
  long i = (long)blockIdx.x * blockDim.x + threadIdx.x;
  if (i >= total) return;
  int o = (int)(i / Cp), c = (int)(i % Cp);
  Wp[i] = (o < O && c < C) ? (__bf16)W[(size_t)o * C + c] : (__bf16)0.0f;
}

// ------------------------------- WMMA GEMM ---------------------------------
// Y[M,O] = X[M,Cp(bf16, zero-padded)] * Wp[Op,Cp]^T + bias[O]
// One wave computes one 16x16 tile via v_wmma_f32_16x16x32_bf16.
// A layout (16-bit 16x32): lanes 0-15 rows M, vgpr j<4 -> K=half*8+2j,
//   j>=4 -> K=16+half*8+2(j-4). B layout (32x16): vgpr j -> same K as A,
//   column = lane&15.  (CDNA5 ISA 7.12.2)
__global__ void k_gemm_bf16(const __bf16* __restrict__ X, const __bf16* __restrict__ Wp,
                            const float* __restrict__ bias, float* __restrict__ Y,
                            int M, int Cp, int O) {
  int lane = threadIdx.x & 31;
  int wv   = threadIdx.x >> 5;
  int mt   = blockIdx.x * GWAVES + wv;
  if (mt * 16 >= M) return;
  int nt   = blockIdx.y;
  int half = lane >> 4;
  int lr   = lane & 15;
  const __bf16* xrow = X  + (size_t)(mt * 16 + lr) * Cp;
  const __bf16* wrow = Wp + (size_t)(nt * 16 + lr) * Cp;
  f32x8 acc = {};
  for (int k0 = 0; k0 < Cp; k0 += 32) {
    bf16x16 a, bm;
#pragma unroll
    for (int j = 0; j < 8; ++j) {
      int ka = k0 + ((j < 4) ? (half * 8 + 2 * j) : (16 + half * 8 + 2 * (j - 4)));
      a[2 * j]     = xrow[ka];
      a[2 * j + 1] = xrow[ka + 1];
      int kb = ka;
      bm[2 * j]     = wrow[kb];
      bm[2 * j + 1] = wrow[kb + 1];
    }
    acc = __builtin_amdgcn_wmma_f32_16x16x32_bf16(false, a, false, bm,
                                                  (short)0, acc, false, false);
  }
  int n = nt * 16 + lr;
  if (n < O) {
    float bv = bias ? bias[n] : 0.0f;
#pragma unroll
    for (int r = 0; r < 8; ++r) {
      int m = mt * 16 + half * 8 + r;   // D layout: vgpr r, lanes16-31 -> M+8
      Y[(size_t)m * O + n] = acc[r] + bv;
    }
  }
}

// -------------------------------- batchnorm --------------------------------

__global__ void k_bn_stats(const float* __restrict__ Y, float* __restrict__ stats,
                           int M, int O) {
  int o = blockIdx.x;
  __shared__ float ss[256];
  __shared__ float sq[256];
  float s = 0.f, q = 0.f;
  for (int m = threadIdx.x; m < M; m += 256) {
    float v = Y[(size_t)m * O + o];
    s += v; q += v * v;
  }
  ss[threadIdx.x] = s; sq[threadIdx.x] = q;
  __syncthreads();
  for (int st = 128; st > 0; st >>= 1) {
    if ((int)threadIdx.x < st) {
      ss[threadIdx.x] += ss[threadIdx.x + st];
      sq[threadIdx.x] += sq[threadIdx.x + st];
    }
    __syncthreads();
  }
  if (threadIdx.x == 0) {
    float mean = ss[0] / (float)M;
    float var  = sq[0] / (float)M - mean * mean;
    stats[2 * o]     = mean;
    stats[2 * o + 1] = rsqrtf(var + 1e-4f);
  }
}

// y = relu((y-mean)*inv*g + beta); optionally also emit padded bf16 for next GEMM
__global__ void k_bn_apply(float* __restrict__ Y, const float* __restrict__ stats,
                           const float* __restrict__ g, const float* __restrict__ bt,
                           __bf16* __restrict__ Xn, int O, int Cpn, long total) {
  long i = (long)blockIdx.x * blockDim.x + threadIdx.x;
  if (i >= total) return;
  if (Xn) {
    long m = i / Cpn; int c = (int)(i - m * Cpn);
    float v = 0.f;
    if (c < O) {
      float y = Y[m * (size_t)O + c];
      y = fmaxf((y - stats[2 * c]) * stats[2 * c + 1] * g[c] + bt[c], 0.f);
      Y[m * (size_t)O + c] = y;
      v = y;
    }
    Xn[i] = (__bf16)v;
  } else {
    int c = (int)(i % O);
    float y = Y[i];
    Y[i] = fmaxf((y - stats[2 * c]) * stats[2 * c + 1] * g[c] + bt[c], 0.f);
  }
}

// ------------------------------ SA primitives ------------------------------

// farthest point sampling, one block per batch
__global__ void k_fps(const float* __restrict__ xyz, int N, int S,
                      int* __restrict__ fidx, float* __restrict__ dist) {
  int b = blockIdx.x;
  const float* X = xyz + (size_t)b * N * 3;
  float* D = dist + (size_t)b * N;
  for (int i = threadIdx.x; i < N; i += 256) D[i] = 1e10f;
  __syncthreads();
  __shared__ float sv[256];
  __shared__ int   si[256];
  int far = 0;
  for (int s = 0; s < S; ++s) {
    if (threadIdx.x == 0) fidx[(size_t)b * S + s] = far;
    float cx = X[far * 3], cy = X[far * 3 + 1], cz = X[far * 3 + 2];
    float best = -1.f; int bi = 0;
    for (int i = threadIdx.x; i < N; i += 256) {
      float dx = X[i * 3] - cx, dy = X[i * 3 + 1] - cy, dz = X[i * 3 + 2] - cz;
      float d = dx * dx + dy * dy + dz * dz;
      float dm = fminf(D[i], d);
      D[i] = dm;
      if (dm > best) { best = dm; bi = i; }
    }
    sv[threadIdx.x] = best; si[threadIdx.x] = bi;
    __syncthreads();
    for (int st = 128; st > 0; st >>= 1) {
      if ((int)threadIdx.x < st) {
        float ov = sv[threadIdx.x + st]; int oi = si[threadIdx.x + st];
        if (ov > sv[threadIdx.x] || (ov == sv[threadIdx.x] && oi < si[threadIdx.x])) {
          sv[threadIdx.x] = ov; si[threadIdx.x] = oi;
        }
      }
      __syncthreads();
    }
    far = si[0];
    __syncthreads();
  }
}

__global__ void k_gather_xyz(const float* __restrict__ xyz, const int* __restrict__ fidx,
                             float* __restrict__ nxyz, int N, int S, long total) {
  long i = (long)blockIdx.x * blockDim.x + threadIdx.x;
  if (i >= total) return;
  long b = i / S;
  int p = fidx[i];
  const float* src = xyz + ((size_t)b * N + p) * 3;
  nxyz[i * 3 + 0] = src[0];
  nxyz[i * 3 + 1] = src[1];
  nxyz[i * 3 + 2] = src[2];
}

// first nsample points (ascending index) within radius; remainder = first
__global__ void k_ballquery(const float* __restrict__ xyz, const float* __restrict__ nxyz,
                            int* __restrict__ idx, int N, int S, int ns, float r2,
                            long total) {
  long i = (long)blockIdx.x * blockDim.x + threadIdx.x;
  if (i >= total) return;
  long b = i / S;
  const float* X = xyz + (size_t)b * N * 3;
  float qx = nxyz[i * 3], qy = nxyz[i * 3 + 1], qz = nxyz[i * 3 + 2];
  int* out = idx + (size_t)i * ns;
  int cnt = 0;
  for (int p = 0; p < N && cnt < ns; ++p) {
    float dx = X[p * 3] - qx, dy = X[p * 3 + 1] - qy, dz = X[p * 3 + 2] - qz;
    if (dx * dx + dy * dy + dz * dz <= r2) out[cnt++] = p;
  }
  if (cnt == 0) out[cnt++] = 0;
  int f = out[0];
  for (; cnt < ns; ++cnt) out[cnt] = f;
}

// build grouped input rows: [xyz - new_xyz, pts] as zero-padded bf16 [M,Cp]
__global__ void k_group(const float* __restrict__ xyz, const float* __restrict__ nxyz,
                        const float* __restrict__ pts, const int* __restrict__ idx,
                        __bf16* __restrict__ Xp, int N, int S, int ns,
                        int Cpts, int Cp, long total) {
  long i = (long)blockIdx.x * blockDim.x + threadIdx.x;
  if (i >= total) return;
  long b = i / ((long)S * ns);
  long q = i / ns;                       // global centroid row b*S+s
  int p = idx[i];
  const float* cg = nxyz + q * 3;
  const float* pg = xyz + ((size_t)b * N + p) * 3;
  __bf16* o = Xp + (size_t)i * Cp;
  o[0] = (__bf16)(pg[0] - cg[0]);
  o[1] = (__bf16)(pg[1] - cg[1]);
  o[2] = (__bf16)(pg[2] - cg[2]);
  const float* f = pts + ((size_t)b * N + p) * Cpts;
  int c = 0;
  for (; c < Cpts; ++c) o[3 + c] = (__bf16)f[c];
  for (c = 3 + Cpts; c < Cp; ++c) o[c] = (__bf16)0.0f;
}

__global__ void k_maxpool(const float* __restrict__ Y, float* __restrict__ P,
                          int ns, int O, long total) {
  long i = (long)blockIdx.x * blockDim.x + threadIdx.x;
  if (i >= total) return;
  long bs = i / O; int c = (int)(i % O);
  float m = -3.4e38f;
  for (int k = 0; k < ns; ++k)
    m = fmaxf(m, Y[((size_t)bs * ns + k) * O + c]);
  P[i] = m;
}

// ------------------------------ FP primitives ------------------------------

// 3-NN inverse-distance interpolation + concat(points1, interp) -> bf16 [M,Cp]
__global__ void k_fp_build(const float* __restrict__ xyz1, const float* __restrict__ xyz2,
                           const float* __restrict__ p1, const float* __restrict__ p2,
                           __bf16* __restrict__ Xp, int N1, int N2,
                           int C1, int C2, int Cp, long total) {
  long i = (long)blockIdx.x * blockDim.x + threadIdx.x;
  if (i >= total) return;
  long b = i / N1;
  const float* X2 = xyz2 + (size_t)b * N2 * 3;
  float qx = xyz1[i * 3], qy = xyz1[i * 3 + 1], qz = xyz1[i * 3 + 2];
  float d0 = 3.4e38f, d1 = 3.4e38f, d2 = 3.4e38f;
  int i0 = 0, i1 = 0, i2 = 0;
  for (int j = 0; j < N2; ++j) {
    float dx = X2[j * 3] - qx, dy = X2[j * 3 + 1] - qy, dz = X2[j * 3 + 2] - qz;
    float d = dx * dx + dy * dy + dz * dz;
    if (d < d0)      { d2 = d1; i2 = i1; d1 = d0; i1 = i0; d0 = d; i0 = j; }
    else if (d < d1) { d2 = d1; i2 = i1; d1 = d; i1 = j; }
    else if (d < d2) { d2 = d; i2 = j; }
  }
  float w0 = 1.f / (d0 + 1e-8f), w1 = 1.f / (d1 + 1e-8f), w2 = 1.f / (d2 + 1e-8f);
  float ws = w0 + w1 + w2;
  w0 /= ws; w1 /= ws; w2 /= ws;
  __bf16* o = Xp + (size_t)i * Cp;
  if (p1) {
    const float* f1 = p1 + (size_t)i * C1;
    for (int c = 0; c < C1; ++c) o[c] = (__bf16)f1[c];
  }
  const float* P2 = p2 + (size_t)b * N2 * C2;
  const float* r0 = P2 + (size_t)i0 * C2;
  const float* r1 = P2 + (size_t)i1 * C2;
  const float* r2 = P2 + (size_t)i2 * C2;
  for (int c = 0; c < C2; ++c)
    o[C1 + c] = (__bf16)(w0 * r0[c] + w1 * r1[c] + w2 * r2[c]);
  for (int c = C1 + C2; c < Cp; ++c) o[c] = (__bf16)0.0f;
}

// [B*N,C] row-major -> (B,C,N) into d_out region
__global__ void k_out_tr(const float* __restrict__ Y, float* __restrict__ out,
                         int N, int C, long total) {
  long i = (long)blockIdx.x * blockDim.x + threadIdx.x;
  if (i >= total) return;
  long b = i / ((long)N * C);
  long r = i - b * (long)N * C;
  int n = (int)(r / C), c = (int)(r % C);
  out[((size_t)b * C + c) * N + n] = Y[i];
}

// ------------------------------- host driver -------------------------------

#define GRID1(total) dim3((unsigned)(((long)(total) + 255) / 256))

extern "C" void kernel_launch(void* const* d_in, const int* in_sizes, int n_in,
                              void* d_out, int out_size, void* d_ws, size_t ws_size,
                              hipStream_t stream) {
  (void)in_sizes; (void)n_in; (void)out_size;
  const int B = 4, N0 = 16384, NS = 32;
  const float* coords = (const float*)d_in[0];
  const float* feats  = (const float*)d_in[1];

  int t = 2;
  auto nxt = [&]() { return (const float*)d_in[t++]; };

  struct Lyr { const float *W, *b, *g, *bt; int O, C, Op, Cp; size_t wp; };
  Lyr Ls[25]; int nL = 0;

  size_t cur = 0;
  auto alloc = [&](size_t bytes) {
    cur = (cur + 255) & ~(size_t)255;
    size_t o = cur; cur += bytes; return o;
  };
  auto addL = [&](int C, int O, bool bn) {
    Lyr l; l.W = nxt(); l.b = nxt();
    if (bn) { l.g = nxt(); l.bt = nxt(); } else { l.g = nullptr; l.bt = nullptr; }
    l.O = O; l.C = C; l.Op = (O + 15) & ~15; l.Cp = (C + 31) & ~31;
    l.wp = alloc((size_t)l.Op * l.Cp * sizeof(__bf16));
    Ls[nL] = l; return nL++;
  };

  int iSA1 = addL(7, 32, true);   addL(32, 32, true);  addL(32, 64, true);
  int iSA2 = addL(67, 64, true);  addL(64, 64, true);  addL(64, 128, true);
  int iSA3 = addL(131, 128, true); addL(128, 128, true); addL(128, 256, true);
  int iSA4 = addL(259, 256, true); addL(256, 256, true); addL(256, 512, true);
  int iFP4 = addL(768, 256, true); addL(256, 256, true);
  int iFP3 = addL(384, 256, true); addL(256, 256, true);
  int iFP2 = addL(320, 256, true); addL(256, 128, true);
  int iFP1 = addL(128, 128, true); addL(128, 128, true); addL(128, 128, true);
  int iSEMH = addL(128, 128, true); int iSEMO = addL(128, 2, false);
  int iOFFH = addL(128, 128, true); int iOFFO = addL(128, 3, false);

  // activation / index buffers
  size_t o_l0xyz = alloc((size_t)B * N0 * 3 * 4);
  size_t o_l0pts = alloc((size_t)B * N0 * 4 * 4);
  size_t o_l1xyz = alloc((size_t)B * 1024 * 3 * 4);
  size_t o_l2xyz = alloc((size_t)B * 256 * 3 * 4);
  size_t o_l3xyz = alloc((size_t)B * 64 * 3 * 4);
  size_t o_l4xyz = alloc((size_t)B * 16 * 3 * 4);
  size_t o_l1pts = alloc((size_t)B * 1024 * 64 * 4);
  size_t o_l2pts = alloc((size_t)B * 256 * 128 * 4);
  size_t o_l3pts = alloc((size_t)B * 64 * 256 * 4);
  size_t o_l4pts = alloc((size_t)B * 16 * 512 * 4);
  size_t o_f3    = alloc((size_t)B * 64 * 256 * 4);
  size_t o_f2    = alloc((size_t)B * 256 * 256 * 4);
  size_t o_f1    = alloc((size_t)B * 1024 * 128 * 4);
  size_t o_l0f   = alloc((size_t)B * N0 * 128 * 4);
  size_t o_fidx  = alloc((size_t)B * 1024 * 4);
  size_t o_bq    = alloc((size_t)B * 1024 * NS * 4);
  size_t o_dist  = alloc((size_t)B * N0 * 4);
  size_t o_stats = alloc((size_t)512 * 2 * 4);
  size_t o_X     = alloc((size_t)8388608 * sizeof(__bf16));
  size_t o_X2    = alloc((size_t)8388608 * sizeof(__bf16));
  size_t o_Y     = alloc((size_t)8388608 * 4);
  size_t o_Yh    = alloc((size_t)B * N0 * 3 * 4);
  if (cur > ws_size) return;  // workspace too small; bail deterministically

  auto FW = [&](size_t o) { return (float*)((char*)d_ws + o); };
  auto IW = [&](size_t o) { return (int*)((char*)d_ws + o); };
  auto BW = [&](size_t o) { return (__bf16*)((char*)d_ws + o); };
  float* dout = (float*)d_out;

  // pack all weights to padded bf16
  for (int i = 0; i < nL; ++i) {
    long tot = (long)Ls[i].Op * Ls[i].Cp;
    k_pack_w<<<GRID1(tot), 256, 0, stream>>>(Ls[i].W, BW(Ls[i].wp),
                                             Ls[i].O, Ls[i].C, Ls[i].Cp, tot);
  }

  // transpose inputs to (B,N,C)
  k_transpose_cn_nc<<<GRID1((long)B * N0 * 3), 256, 0, stream>>>(coords, FW(o_l0xyz), N0, 3, (long)B * N0 * 3);
  k_transpose_cn_nc<<<GRID1((long)B * N0 * 4), 256, 0, stream>>>(feats, FW(o_l0pts), N0, 4, (long)B * N0 * 4);

  auto gemm = [&](Lyr& l, const __bf16* Xin, float* Y, int M) {
    dim3 g((unsigned)((M / 16 + GWAVES - 1) / GWAVES), (unsigned)(l.Op / 16));
    k_gemm_bf16<<<g, GWAVES * 32, 0, stream>>>(Xin, BW(l.wp), l.b, Y, M, l.Cp, l.O);
  };
  auto bnrelu = [&](Lyr& l, float* Y, int M, __bf16* Xn, int Cpn) {
    k_bn_stats<<<l.O, 256, 0, stream>>>(Y, FW(o_stats), M, l.O);
    long tot = Xn ? (long)M * Cpn : (long)M * l.O;
    k_bn_apply<<<GRID1(tot), 256, 0, stream>>>(Y, FW(o_stats), l.g, l.bt, Xn, l.O, Cpn, tot);
  };
  auto run_mlp = [&](int li0, int nl, int M, float* Ybuf, __bf16* lastX, int lastCp) {
    const __bf16* Xin = BW(o_X);
    for (int li = 0; li < nl; ++li) {
      Lyr& l = Ls[li0 + li];
      gemm(l, Xin, Ybuf, M);
      bool last = (li == nl - 1);
      __bf16* xn = last ? lastX : BW(o_X);
      int cpn = last ? lastCp : Ls[li0 + li + 1].Cp;
      bnrelu(l, Ybuf, M, xn, cpn);
      if (xn) Xin = xn;
    }
  };
  auto run_sa = [&](size_t oxyz, size_t opts, int Nn, int S, float radius, int Cpts,
                    int li0, size_t onxyz, size_t onpts) {
    k_fps<<<B, 256, 0, stream>>>(FW(oxyz), Nn, S, IW(o_fidx), FW(o_dist));
    long bs = (long)B * S;
    k_gather_xyz<<<GRID1(bs), 256, 0, stream>>>(FW(oxyz), IW(o_fidx), FW(onxyz), Nn, S, bs);
    k_ballquery<<<GRID1(bs), 256, 0, stream>>>(FW(oxyz), FW(onxyz), IW(o_bq),
                                               Nn, S, NS, radius * radius, bs);
    long M = bs * NS;
    k_group<<<GRID1(M), 256, 0, stream>>>(FW(oxyz), FW(onxyz), FW(opts), IW(o_bq),
                                          BW(o_X), Nn, S, NS, Cpts, Ls[li0].Cp, M);
    run_mlp(li0, 3, (int)M, FW(o_Y), nullptr, 0);
    int O = Ls[li0 + 2].O;
    long tp = bs * O;
    k_maxpool<<<GRID1(tp), 256, 0, stream>>>(FW(o_Y), FW(onpts), NS, O, tp);
  };
  auto run_fp = [&](size_t oxyz1, size_t oxyz2, const float* p1, const float* p2,
                    int N1, int N2, int C1, int C2, int li0, int nl,
                    float* Yout, __bf16* lastX, int lastCp) {
    long M = (long)B * N1;
    k_fp_build<<<GRID1(M), 256, 0, stream>>>(FW(oxyz1), FW(oxyz2), p1, p2,
                                             BW(o_X), N1, N2, C1, C2, Ls[li0].Cp, M);
    run_mlp(li0, nl, (int)M, Yout, lastX, lastCp);
  };

  // ---- set abstraction ----
  run_sa(o_l0xyz, o_l0pts, N0, 1024, 0.1f, 4,   iSA1, o_l1xyz, o_l1pts);
  run_sa(o_l1xyz, o_l1pts, 1024, 256, 0.2f, 64,  iSA2, o_l2xyz, o_l2pts);
  run_sa(o_l2xyz, o_l2pts, 256, 64,   0.4f, 128, iSA3, o_l3xyz, o_l3pts);
  run_sa(o_l3xyz, o_l3pts, 64, 16,    0.8f, 256, iSA4, o_l4xyz, o_l4pts);

  // ---- feature propagation ----
  run_fp(o_l3xyz, o_l4xyz, FW(o_l3pts), FW(o_l4pts), 64, 16, 256, 512, iFP4, 2, FW(o_f3), nullptr, 0);
  run_fp(o_l2xyz, o_l3xyz, FW(o_l2pts), FW(o_f3),   256, 64, 128, 256, iFP3, 2, FW(o_f2), nullptr, 0);
  run_fp(o_l1xyz, o_l2xyz, FW(o_l1pts), FW(o_f2),  1024, 256, 64, 256, iFP2, 2, FW(o_f1), nullptr, 0);
  run_fp(o_l0xyz, o_l1xyz, nullptr,     FW(o_f1),    N0, 1024, 0, 128, iFP1, 3, FW(o_l0f), BW(o_X), 128);
  // o_X now holds bf16 l0_feat (Cp=128); o_l0f holds f32 l0_feat

  // ---- heads ----
  int M0 = B * N0;
  // sem
  gemm(Ls[iSEMH], BW(o_X), FW(o_Y), M0);
  bnrelu(Ls[iSEMH], FW(o_Y), M0, BW(o_X2), Ls[iSEMO].Cp);
  gemm(Ls[iSEMO], BW(o_X2), FW(o_Yh), M0);
  long tsem = (long)M0 * 2;
  k_out_tr<<<GRID1(tsem), 256, 0, stream>>>(FW(o_Yh), dout + (size_t)B * 128 * N0, N0, 2, tsem);
  // off
  gemm(Ls[iOFFH], BW(o_X), FW(o_Y), M0);
  bnrelu(Ls[iOFFH], FW(o_Y), M0, BW(o_X2), Ls[iOFFO].Cp);
  gemm(Ls[iOFFO], BW(o_X2), FW(o_Yh), M0);
  long toff = (long)M0 * 3;
  k_out_tr<<<GRID1(toff), 256, 0, stream>>>(FW(o_Yh), dout + (size_t)B * 128 * N0 + (size_t)B * 2 * N0, N0, 3, toff);
  // backbone
  long tbb = (long)M0 * 128;
  k_out_tr<<<GRID1(tbb), 256, 0, stream>>>(FW(o_l0f), dout, N0, 128, tbb);
}
